// RNN1_47021301957328
// MI455X (gfx1250) — matrix operand order repaired and checked by reading, not compiled, not run on hardware
//
#include <hip/hip_runtime.h>

// ---------------------------------------------------------------------------
// RNN (embed -> fc1 -> x2h -> 128x tanh-RNN -> fc2) for MI455X (gfx1250).
// Matrix math on v_wmma_f32_16x16x32_f16 (wave32), fp32 accumulate.
// fc1 and x2h are algebraically fused: Wc = x2h_w @ fc1_w  ([512,128]),
// bstep = x2h_b + x2h_w@fc1_b + h2h_b, so each RNN step is
//   H' = tanh(feats_t @ Wc^T + H @ h2h^T + bstep)
// Step kernel stages its A tile in LDS via global_load_async_to_lds_b128
// (ASYNCcnt) and reads A fragments with ds_load_b128; B comes from L2.
// ---------------------------------------------------------------------------

typedef __attribute__((ext_vector_type(16))) _Float16 v16h;
typedef __attribute__((ext_vector_type(8)))  _Float16 v8h;
typedef __attribute__((ext_vector_type(4)))  _Float16 v4h;
typedef __attribute__((ext_vector_type(8)))  float    v8f;
typedef __attribute__((ext_vector_type(4)))  float    v4f;

#define BB      1024      // batch
#define TT      128       // timesteps
#define RNNIN   128       // fused input width (fc1 out == 128)
#define HID     512
#define LABELS  100
#define NPAD    112       // fc2 rows padded to a multiple of 16
#define KTOT    (RNNIN + HID)   // 640 combined K per step

// ----- WMMA fragment loaders (documented CDNA5 wave32 layouts) -------------
// A 16x32 f16 (MxK): lane m (0-15) holds K=kb..kb+7 in v[0..7] and
// K=kb+16..kb+23 in v[8..15], kb = 0 for lanes 0-15, 8 for lanes 16-31.
// Works for global or LDS base pointers (LDS -> ds_load_b128).
__device__ __forceinline__ v16h load_fragA(const _Float16* __restrict__ base,
                                           int lda, int k0) {
  int lane = threadIdx.x & 31;
  int m    = lane & 15;
  int kb   = (lane >> 4) << 3;                  // 0 or 8
  const _Float16* p = base + (size_t)m * lda + k0 + kb;
  v8h lo = *(const v8h*)(p);
  v8h hi = *(const v8h*)(p + 16);
  v16h r;
#pragma unroll
  for (int i = 0; i < 8; ++i) { r[i] = lo[i]; r[i + 8] = hi[i]; }
  return r;
}

// B 32x16 f16 (KxN) with B = W^T, W row-major [N,K]: lane n (0-15) holds
// B[k][n] = W[n0+n][k] for k = k0+kb..+7 in v[0..7], k0+kb+16..+23 in v[8..15], kb = 0 / 8.
__device__ __forceinline__ v16h load_fragB(const _Float16* __restrict__ w,
                                           int ldw, int n0, int k0) {
  int lane = threadIdx.x & 31;
  int n    = lane & 15;
  int kb   = (lane >> 4) << 3;                  // 0 or 8
  const _Float16* p = w + (size_t)(n0 + n) * ldw + k0 + kb;
  v8h lo = *(const v8h*)(p);
  v8h hi = *(const v8h*)(p + 16);
  v16h r;
#pragma unroll
  for (int i = 0; i < 8; ++i) { r[i] = lo[i]; r[i + 8] = hi[i]; }
  return r;
}

// Async copy of one 16-byte chunk from global to LDS (CDNA5 async path).
__device__ __forceinline__ void async_copy_b128(unsigned lds_byte_addr,
                                                const _Float16* gptr) {
  unsigned long long ga = (unsigned long long)(const void*)gptr;
  asm volatile("global_load_async_to_lds_b128 %0, %1, off"
               :: "v"(lds_byte_addr), "v"(ga)
               : "memory");
}

__device__ __forceinline__ void wait_asynccnt0() {
  asm volatile("s_wait_asynccnt 0x0" ::: "memory");
}

// ----- prep: f32 weights -> f16, zero h0, zero-pad fc2 ---------------------
__global__ __launch_bounds__(256) void prep_convert(
    const float* __restrict__ h2h_w, const float* __restrict__ fc2_w,
    _Float16* __restrict__ h2h_f16, _Float16* __restrict__ fc2_f16,
    _Float16* __restrict__ h0) {
  int i = blockIdx.x * blockDim.x + threadIdx.x;
  if (i < HID * HID) {
    h2h_f16[i] = (_Float16)h2h_w[i];
    return;
  }
  int j = i - HID * HID;
  if (j < NPAD * HID) {
    int row = j / HID;
    fc2_f16[j] = (row < LABELS) ? (_Float16)fc2_w[j] : (_Float16)0.0f;
    return;
  }
  int k = j - NPAD * HID;
  if (k < BB * HID) h0[k] = (_Float16)0.0f;
}

// Wc[o][k] = sum_n x2h_w[o][n] * fc1_w[n][k]   (512x128, one thread each)
__global__ __launch_bounds__(256) void prep_wc(
    const float* __restrict__ x2h_w, const float* __restrict__ fc1_w,
    _Float16* __restrict__ wc) {
  int i = blockIdx.x * blockDim.x + threadIdx.x;
  if (i >= HID * RNNIN) return;
  int o = i >> 7, k = i & 127;
  float s = 0.0f;
#pragma unroll 8
  for (int n = 0; n < RNNIN; ++n)
    s = fmaf(x2h_w[o * RNNIN + n], fc1_w[n * RNNIN + k], s);
  wc[i] = (_Float16)s;
}

// bstep[o] = x2h_b[o] + h2h_b[o] + sum_n x2h_w[o][n]*fc1_b[n]
__global__ __launch_bounds__(256) void prep_bias(
    const float* __restrict__ x2h_w, const float* __restrict__ fc1_b,
    const float* __restrict__ x2h_b, const float* __restrict__ h2h_b,
    float* __restrict__ bstep) {
  int o = blockIdx.x * blockDim.x + threadIdx.x;
  if (o >= HID) return;
  float s = x2h_b[o] + h2h_b[o];
#pragma unroll 8
  for (int n = 0; n < RNNIN; ++n) s = fmaf(x2h_w[o * RNNIN + n], fc1_b[n], s);
  bstep[o] = s;
}

// ----- embedding gather -> f16 feats, layout [T, B, 128] -------------------
__global__ __launch_bounds__(256) void embed_kernel(
    const int* __restrict__ x, const float* __restrict__ len_emb,
    const float* __restrict__ ipd_emb, _Float16* __restrict__ feats) {
  size_t tid = (size_t)blockIdx.x * blockDim.x + threadIdx.x;
  size_t e = tid * 4;                        // 4 elements per thread
  if (e >= (size_t)BB * TT * RNNIN) return;
  size_t row = e >> 7;                       // row = t*B + b
  int k = (int)(e & 127);
  int t = (int)(row >> 10);
  int b = (int)(row & 1023);
  int which = (k >= 64);
  int idx = x[((size_t)b * TT + t) * 2 + which];
  const float* src = which ? (ipd_emb + (size_t)idx * 64 + (k - 64))
                           : (len_emb + (size_t)idx * 64 + k);
  v4f v = *(const v4f*)src;
  v4h o;
#pragma unroll
  for (int i = 0; i < 4; ++i) o[i] = (_Float16)v[i];
  *(v4h*)(feats + e) = o;
}

// ----- one RNN step: H' = tanh(feats_t @ Wc^T + H @ h2h^T + bstep) ---------
// grid (64, 2), block 256 (8 waves). Block owns one 16-row M tile; its A
// data (16 x 640 f16 = 20KB) is staged in LDS with async loads. Each wave
// computes a 16x32 output slab (two 16x16 N tiles, A frag reused), 40 WMMAs.
__global__ __launch_bounds__(256) void rnn_step_kernel(
    const _Float16* __restrict__ feats,      // [T*B, 128]
    const _Float16* __restrict__ wc,         // [512, 128]
    const _Float16* __restrict__ h2h,        // [512, 512]
    const float*    __restrict__ bstep,      // [512]
    const _Float16* __restrict__ h_in,       // [1024, 512]
    _Float16*       __restrict__ h_out,      // [1024, 512]
    int t) {
  // A tile in LDS: 16 rows x 640 cols (cols 0..127 = feats, 128..639 = H)
  __shared__ _Float16 smA[16 * KTOT];

  int wave = threadIdx.x >> 5;
  int lane = threadIdx.x & 31;
  int mt = blockIdx.x;                               // M tile (16 rows)
  int n0 = blockIdx.y * 256 + wave * 32;             // first of two N tiles

  const _Float16* a_feats = feats + ((size_t)t * BB + (size_t)mt * 16) * RNNIN;
  const _Float16* a_h     = h_in + (size_t)mt * 16 * HID;
  unsigned sm_base = (unsigned)(unsigned long long)(const void*)smA;

  // Cooperative async stage: 1280 16B chunks (16 rows x 80), 5 per thread.
#pragma unroll
  for (int it = 0; it < 5; ++it) {
    int ch  = threadIdx.x + it * 256;
    int row = ch / 80;
    int q   = ch % 80;                               // 8-half chunk in row
    const _Float16* g;
    unsigned loff;
    if (q < 16) {                                    // feats portion
      g    = a_feats + row * RNNIN + q * 8;
      loff = (unsigned)(row * KTOT + q * 8) * 2u;
    } else {                                         // H portion
      int qq = q - 16;
      g    = a_h + row * HID + qq * 8;
      loff = (unsigned)(row * KTOT + RNNIN + qq * 8) * 2u;
    }
    async_copy_b128(sm_base + loff, g);
  }
  wait_asynccnt0();
  __syncthreads();

  v8f acc0 = {}, acc1 = {};                          // two N-tile chains
#pragma unroll
  for (int k0 = 0; k0 < RNNIN; k0 += 32) {           // x part (K=128, from LDS)
    v16h a  = load_fragA(smA, KTOT, k0);
    v16h b0 = load_fragB(wc, RNNIN, n0, k0);
    v16h b1 = load_fragB(wc, RNNIN, n0 + 16, k0);
    acc0 = __builtin_amdgcn_wmma_f32_16x16x32_f16(false, a, false, b0,
                                                  (short)0, acc0, false, false);
    acc1 = __builtin_amdgcn_wmma_f32_16x16x32_f16(false, a, false, b1,
                                                  (short)0, acc1, false, false);
  }
#pragma unroll
  for (int k0 = 0; k0 < HID; k0 += 32) {             // h part (K=512, from LDS)
    v16h a  = load_fragA(smA + RNNIN, KTOT, k0);
    v16h b0 = load_fragB(h2h, HID, n0, k0);
    v16h b1 = load_fragB(h2h, HID, n0 + 16, k0);
    acc0 = __builtin_amdgcn_wmma_f32_16x16x32_f16(false, a, false, b0,
                                                  (short)0, acc0, false, false);
    acc1 = __builtin_amdgcn_wmma_f32_16x16x32_f16(false, a, false, b1,
                                                  (short)0, acc1, false, false);
  }

  // C/D layout: VGPR r -> M = r (lanes 0-15) or r+8 (lanes 16-31), N = lane&15
  int col0 = n0 + (lane & 15);
  float bias0 = bstep[col0];
  float bias1 = bstep[col0 + 16];
  int rbase = mt * 16 + ((lane >> 4) << 3);
#pragma unroll
  for (int r = 0; r < 8; ++r) {
    size_t rowoff = (size_t)(rbase + r) * HID;
    h_out[rowoff + col0]      = (_Float16)tanhf(acc0[r] + bias0);
    h_out[rowoff + col0 + 16] = (_Float16)tanhf(acc1[r] + bias1);
  }
}

// ----- fc2: out = H_last @ fc2_w^T + fc2_b ---------------------------------
// grid 64, block 224 (7 waves -> N tiles 0..6, rows padded to 112).
__global__ __launch_bounds__(224) void fc2_kernel(
    const _Float16* __restrict__ h,          // [1024, 512]
    const _Float16* __restrict__ w,          // [112, 512] zero-padded
    const float* __restrict__ bias,          // [100]
    float* __restrict__ out) {               // [1024, 100]
  int wave = threadIdx.x >> 5;
  int lane = threadIdx.x & 31;
  int mt = blockIdx.x;
  int n0 = wave * 16;
  const _Float16* a = h + (size_t)mt * 16 * HID;

  v8f acc0 = {}, acc1 = {};
#pragma unroll
  for (int k0 = 0; k0 < HID; k0 += 64) {
    v16h av = load_fragA(a, HID, k0);
    v16h bv = load_fragB(w, HID, n0, k0);
    acc0 = __builtin_amdgcn_wmma_f32_16x16x32_f16(false, av, false, bv,
                                                  (short)0, acc0, false, false);
    av = load_fragA(a, HID, k0 + 32);
    bv = load_fragB(w, HID, n0, k0 + 32);
    acc1 = __builtin_amdgcn_wmma_f32_16x16x32_f16(false, av, false, bv,
                                                  (short)0, acc1, false, false);
  }
  int col = n0 + (lane & 15);
  if (col < LABELS) {
    float bv = bias[col];
    int rbase = mt * 16 + ((lane >> 4) << 3);
#pragma unroll
    for (int r = 0; r < 8; ++r)
      out[(size_t)(rbase + r) * LABELS + col] = acc0[r] + acc1[r] + bv;
  }
}

// ---------------------------------------------------------------------------
extern "C" void kernel_launch(void* const* d_in, const int* in_sizes, int n_in,
                              void* d_out, int out_size, void* d_ws,
                              size_t ws_size, hipStream_t stream) {
  const int*   x       = (const int*)d_in[0];
  const float* len_emb = (const float*)d_in[1];
  const float* ipd_emb = (const float*)d_in[2];
  const float* fc1_w   = (const float*)d_in[3];
  const float* fc1_b   = (const float*)d_in[4];
  const float* x2h_w   = (const float*)d_in[5];
  const float* x2h_b   = (const float*)d_in[6];
  const float* h2h_w   = (const float*)d_in[7];
  const float* h2h_b   = (const float*)d_in[8];
  const float* fc2_w   = (const float*)d_in[9];
  const float* fc2_b   = (const float*)d_in[10];
  float* out = (float*)d_out;

  // workspace carve-up (256B-aligned), ~36.5 MB total
  char* ws = (char*)d_ws;
  size_t off = 0;
  auto carve = [&](size_t bytes) -> void* {
    void* p = ws + off;
    off = (off + bytes + 255) & ~(size_t)255;
    return p;
  };
  _Float16* h2h_f16 = (_Float16*)carve((size_t)HID * HID * 2);
  _Float16* wc_f16  = (_Float16*)carve((size_t)HID * RNNIN * 2);
  _Float16* fc2_f16 = (_Float16*)carve((size_t)NPAD * HID * 2);
  float*    bstep   = (float*)carve((size_t)HID * 4);
  _Float16* hbuf0   = (_Float16*)carve((size_t)BB * HID * 2);
  _Float16* hbuf1   = (_Float16*)carve((size_t)BB * HID * 2);
  _Float16* feats   = (_Float16*)carve((size_t)BB * TT * RNNIN * 2);

  // prep: weight conversion, fused weight/bias, h0 = 0
  int prep_elems = HID * HID + NPAD * HID + BB * HID;
  prep_convert<<<(prep_elems + 255) / 256, 256, 0, stream>>>(
      h2h_w, fc2_w, h2h_f16, fc2_f16, hbuf0);
  prep_wc<<<(HID * RNNIN + 255) / 256, 256, 0, stream>>>(x2h_w, fc1_w, wc_f16);
  prep_bias<<<(HID + 255) / 256, 256, 0, stream>>>(x2h_w, fc1_b, x2h_b, h2h_b,
                                                   bstep);

  // embedding gather -> f16 feats [T, B, 128]
  size_t emb_threads = (size_t)BB * TT * RNNIN / 4;
  embed_kernel<<<(unsigned)((emb_threads + 255) / 256), 256, 0, stream>>>(
      x, len_emb, ipd_emb, feats);

  // 128 sequential RNN steps, double-buffered H (t=127 writes hbuf0)
  _Float16* bufs[2] = {hbuf0, hbuf1};
  for (int t = 0; t < TT; ++t) {
    rnn_step_kernel<<<dim3(64, 2), 256, 0, stream>>>(
        feats, wc_f16, h2h_f16, bstep, bufs[t & 1], bufs[(t + 1) & 1], t);
  }

  // fc2 on final hidden state
  fc2_kernel<<<64, 224, 0, stream>>>(hbuf0, fc2_f16, fc2_b, out);
}
